// ReferentialV2_68728066670878
// MI455X (gfx1250) — compile-verified, matrix operand order checked by reading, not run on hardware
//
#include <hip/hip_runtime.h>
#include <hip/hip_bf16.h>
#include <math.h>

typedef _Float16 h16;
typedef __attribute__((ext_vector_type(16))) _Float16 v16h;
typedef __attribute__((ext_vector_type(8)))  float    v8f;

#define NB      128
#define KPH     3
#define MAXPER  48
#define MAXTOK  144
#define HD      512
#define NHEADS  8
#define DHEAD   64
#define NMEM    8
#define VOC     8000
#define TSLOT   145   // MAX_TOK + 1 cache slots
#define LAT     128
#define EOS_TOK 2
#define BOS_TOK 1

// ---------------------------------------------------------------------------
// WMMA fragment helpers (16x16x32 f16 -> f32)
// A (16xK, row-major, f16): lane row = lane&15, K packing: halves 0..7 -> kb+0..7,
// halves 8..15 -> kb+16..23, kb = ((lane>>4)<<3). B stored as W^T rows [N][K] so
// the same loader works (row = output column n).
// ---------------------------------------------------------------------------
static __device__ __forceinline__ v16h ld_frag(const h16* p0, int ld, int lane) {
  const h16* p = p0 + (size_t)(lane & 15) * ld + ((lane >> 4) << 3);
  v16h f;
#pragma unroll
  for (int i = 0; i < 8; ++i) f[i] = p[i];
#pragma unroll
  for (int i = 0; i < 8; ++i) f[8 + i] = p[16 + i];
  return f;
}

static __device__ __forceinline__ void st_tile(float* out, int ld, int lane, v8f a) {
  int n = lane & 15, mb = (lane >> 4) << 3;
#pragma unroll
  for (int r = 0; r < 8; ++r) out[(size_t)(mb + r) * ld + n] = a[r];
}

static __device__ __forceinline__ v8f ld_tile(const float* in, int ld, int lane) {
  int n = lane & 15, mb = (lane >> 4) << 3;
  v8f a;
#pragma unroll
  for (int r = 0; r < 8; ++r) a[r] = in[(size_t)(mb + r) * ld + n];
  return a;
}

// OUT[16][N] = A[16][K] (f16) * Wt[N][K] (f16, transposed weight) (+ C)
// Waves split the N/16 tiles; all lanes of a wave follow the same path (EXEC full).
static __device__ void gemm16(const h16* __restrict__ A, int lda, int K,
                              const h16* __restrict__ Wt, int ldw, int N,
                              float* __restrict__ out, int ldo,
                              const float* __restrict__ Cb, int ldc,
                              int wave, int nwaves, int lane) {
  for (int nt = wave; nt * 16 < N; nt += nwaves) {
    int n0 = nt * 16;
    v8f acc;
    if (Cb) acc = ld_tile(Cb + n0, ldc, lane);
    else    acc = (v8f){0.f, 0.f, 0.f, 0.f, 0.f, 0.f, 0.f, 0.f};
    for (int k0 = 0; k0 < K; k0 += 32) {
      v16h a = ld_frag(A + k0, lda, lane);
      v16h b = ld_frag(Wt + (size_t)n0 * ldw + k0, ldw, lane);
      acc = __builtin_amdgcn_wmma_f32_16x16x32_f16(false, a, false, b,
                                                   (short)0, acc, false, false);
    }
    st_tile(out + n0, ldo, lane, acc);
  }
}

static __device__ __forceinline__ float gelu_f(float x) {
  const float c = 0.7978845608028654f;                 // sqrt(2/pi)
  float t = c * (x + 0.044715f * x * x * x);
  return 0.5f * x * (1.f + tanhf(t));
}

// ---------------------------------------------------------------------------
// Prepass kernels
// ---------------------------------------------------------------------------
__global__ void zero_kernel(float* p, int n) {
  int i = blockIdx.x * blockDim.x + threadIdx.x;
  if (i < n) p[i] = 0.f;
}

// fp32 [K][N] -> f16 [N][K]
__global__ void transpose_f16(const float* __restrict__ in, h16* __restrict__ out,
                              int K, int N) {
  size_t i = (size_t)blockIdx.x * blockDim.x + threadIdx.x;
  if (i >= (size_t)K * N) return;
  int n = (int)(i % N), k = (int)(i / N);
  out[(size_t)n * K + k] = (h16)in[i];
}

// memories[bk][j] = zw[bk] * sum_l z[bk][l] * W_l2d[l][j]   (f16 out, [384][4096])
__global__ void mem_kernel(const float* __restrict__ z, const float* __restrict__ zw,
                           const float* __restrict__ Wl2d, h16* __restrict__ out) {
  size_t i = (size_t)blockIdx.x * blockDim.x + threadIdx.x;
  if (i >= (size_t)NB * KPH * NMEM * HD) return;
  int j  = (int)(i & (NMEM * HD - 1));
  int bk = (int)(i >> 12);
  const float* zr = z + (size_t)bk * LAT;
  float s = 0.f;
  for (int l = 0; l < LAT; ++l) s += zr[l] * Wl2d[(size_t)l * (NMEM * HD) + j];
  out[i] = (h16)(s * zw[bk]);
}

// ckfull/cvfull[3072][512] = memories[3072][512] @ Wck/Wcv  (WMMA prepass GEMM)
__global__ __launch_bounds__(256) void ckv_kernel(const h16* __restrict__ A,
                                                  const h16* __restrict__ WckT,
                                                  const h16* __restrict__ WcvT,
                                                  float* __restrict__ ck,
                                                  float* __restrict__ cv) {
  int mt = blockIdx.x;  // 0..191 (3072/16 M-tiles)
  const h16* Wt = blockIdx.y ? WcvT : WckT;
  float* out = (blockIdx.y ? cv : ck) + (size_t)mt * 16 * HD;
  const h16* Ab = A + (size_t)mt * 16 * HD;
  int tid = threadIdx.x, wave = tid >> 5, lane = tid & 31;
  gemm16(Ab, HD, HD, Wt, HD, HD, out, HD, nullptr, 0, wave, 8, lane);
}

// ---------------------------------------------------------------------------
// Persistent decode kernel: 8 blocks x 512 threads, each block owns 16 rows
// ---------------------------------------------------------------------------
struct Params {
  const float* z_weights;
  const float* token_emb;
  const h16 *Wq, *Wk, *Wv, *Wo, *Wcq, *Wco, *W1, *W2, *Wout;
  const float *ckf, *cvf;
  h16 *kc, *vc;
  float *out_tok, *out_gen, *out_bnd, *out_hid;
};

__global__ __launch_bounds__(512) void decode_kernel(Params P) {
  __shared__ __attribute__((aligned(16))) float X32[16 * HD];
  __shared__ __attribute__((aligned(16))) float Q32[16 * HD];
  __shared__ __attribute__((aligned(16))) float T32[16 * HD];
  __shared__ __attribute__((aligned(16))) h16   A16a[16 * HD];
  __shared__ __attribute__((aligned(16))) h16   A16b[16 * HD];
  __shared__ int   st_cur[16], st_tip[16], st_pos[16], st_done[16];
  __shared__ int   st_act[16 * KPH], st_bounds[16 * KPH], st_nxt[16], st_mi[16];
  __shared__ float st_mg[16];

  const int tid  = threadIdx.x;
  const int wave = tid >> 5;
  const int lane = tid & 31;
  const int b0   = blockIdx.x * 16;
  const float scale = 0.125f;  // 1/sqrt(64)

  auto to_f16 = [&](const float* src, h16* dst) {
    for (int i = tid; i < 16 * HD; i += 512) dst[i] = (h16)src[i];
  };

  // ---- one attn_step for the block's 16 rows; kv slot = step ----
  auto attn_step = [&](int step) {
    to_f16(X32, A16a);
    __syncthreads();
    // q, k
    gemm16(A16a, HD, HD, P.Wq, HD, HD, Q32, HD, nullptr, 0, wave, 16, lane);
    gemm16(A16a, HD, HD, P.Wk, HD, HD, T32, HD, nullptr, 0, wave, 16, lane);
    __syncthreads();
    for (int i = tid; i < 16 * HD; i += 512) {
      int r = i >> 9, c = i & (HD - 1);
      P.kc[((size_t)(b0 + r) * TSLOT + step) * HD + c] = (h16)T32[i];
    }
    __syncthreads();
    // v
    gemm16(A16a, HD, HD, P.Wv, HD, HD, T32, HD, nullptr, 0, wave, 16, lane);
    __syncthreads();
    for (int i = tid; i < 16 * HD; i += 512) {
      int r = i >> 9, c = i & (HD - 1);
      P.vc[((size_t)(b0 + r) * TSLOT + step) * HD + c] = (h16)T32[i];
    }
    __syncthreads();
    // self attention (online softmax), thread per (row, head) -> T32
    if (tid < 128) {
      int row = tid >> 3, h = tid & 7;
      const float* q = &Q32[row * HD + h * DHEAD];
      const h16* kcb = P.kc + (size_t)(b0 + row) * TSLOT * HD + h * DHEAD;
      const h16* vcb = P.vc + (size_t)(b0 + row) * TSLOT * HD + h * DHEAD;
      float mx = -3.0e38f, l = 0.f;
      float acc[DHEAD];
#pragma unroll
      for (int d = 0; d < DHEAD; ++d) acc[d] = 0.f;
      for (int t = 0; t <= step; ++t) {
        const h16* kp = kcb + (size_t)t * HD;
        float s = 0.f;
#pragma unroll
        for (int d = 0; d < DHEAD; ++d) s += q[d] * (float)kp[d];
        s *= scale;
        float mnew = fmaxf(mx, s);
        float corr = expf(mx - mnew);
        float w    = expf(s - mnew);
        l = l * corr + w;
        const h16* vp = vcb + (size_t)t * HD;
#pragma unroll
        for (int d = 0; d < DHEAD; ++d) acc[d] = acc[d] * corr + w * (float)vp[d];
        mx = mnew;
      }
      float inv = 1.f / l;
#pragma unroll
      for (int d = 0; d < DHEAD; ++d) T32[row * HD + h * DHEAD + d] = acc[d] * inv;
    }
    __syncthreads();
    to_f16(T32, A16b);
    __syncthreads();
    gemm16(A16b, HD, HD, P.Wo, HD, HD, X32, HD, X32, HD, wave, 16, lane);  // x += sa@Wo
    __syncthreads();
    // cross attention
    to_f16(X32, A16a);
    __syncthreads();
    gemm16(A16a, HD, HD, P.Wcq, HD, HD, Q32, HD, nullptr, 0, wave, 16, lane);
    __syncthreads();
    if (tid < 128) {
      int row = tid >> 3, h = tid & 7;
      int b = b0 + row;
      int i = st_mi[row];
      float g = st_mg[row];
      const float* cq = &Q32[row * HD + h * DHEAD];
      size_t base = (((size_t)b * KPH + i) * NMEM) * HD + h * DHEAD;
      float sc[NMEM];
#pragma unroll
      for (int j = 0; j < NMEM; ++j) {
        const float* kp = P.ckf + base + (size_t)j * HD;
        float s = 0.f;
#pragma unroll
        for (int d = 0; d < DHEAD; ++d) s += cq[d] * kp[d];
        sc[j] = s * scale * g;
      }
      float mx = sc[0];
#pragma unroll
      for (int j = 1; j < NMEM; ++j) mx = fmaxf(mx, sc[j]);
      float l = 0.f;
#pragma unroll
      for (int j = 0; j < NMEM; ++j) { sc[j] = expf(sc[j] - mx); l += sc[j]; }
      float inv = 1.f / l;
#pragma unroll
      for (int d = 0; d < DHEAD; ++d) {
        float a = 0.f;
#pragma unroll
        for (int j = 0; j < NMEM; ++j)
          a += sc[j] * P.cvf[base + (size_t)j * HD + d];
        T32[row * HD + h * DHEAD + d] = a * inv * g;
      }
    }
    __syncthreads();
    to_f16(T32, A16b);
    __syncthreads();
    gemm16(A16b, HD, HD, P.Wco, HD, HD, X32, HD, X32, HD, wave, 16, lane);  // x += ca@Wco
    __syncthreads();
    // FFN: x += gelu(x@W1)@W2, K-chunked by 512 to bound LDS
    to_f16(X32, A16a);
    __syncthreads();
    for (int ch = 0; ch < 4; ++ch) {
      gemm16(A16a, HD, HD, P.W1 + (size_t)ch * HD * HD, HD, HD,
             T32, HD, nullptr, 0, wave, 16, lane);
      __syncthreads();
      for (int i = tid; i < 16 * HD; i += 512) A16b[i] = (h16)gelu_f(T32[i]);
      __syncthreads();
      gemm16(A16b, HD, HD, P.W2 + (size_t)ch * HD, 4 * HD, HD,
             X32, HD, X32, HD, wave, 16, lane);
      __syncthreads();
    }
  };

  // ---- init per-row decode state ----
  if (tid < 16) {
    int r = tid;
    for (int k = 0; k < KPH; ++k) {
      st_act[r * KPH + k] = (P.z_weights[(size_t)(b0 + r) * KPH + k] > 0.01f) ? 1 : 0;
      st_bounds[r * KPH + k] = 0;
    }
    st_cur[r] = 0; st_tip[r] = 0; st_pos[r] = 0;
    st_done[r] = st_act[r * KPH] ? 0 : 1;
    st_mi[r] = 0;
    st_mg[r] = (st_act[r * KPH] && !st_done[r]) ? 1.f : 0.f;
  }
  for (int i = tid; i < 16 * HD; i += 512)
    X32[i] = P.token_emb[(size_t)BOS_TOK * HD + (i & (HD - 1))];
  __syncthreads();

  attn_step(0);

  for (int t = 0; t < MAXTOK; ++t) {
    // hiddens[t] = x (post attn_step)
    for (int i = tid; i < 16 * HD; i += 512) {
      int r = i >> 9, c = i & (HD - 1);
      P.out_hid[((size_t)t * NB + b0 + r) * HD + c] = X32[i];
    }
    // logits = x @ Wout fused with per-lane argmax
    to_f16(X32, A16a);
    __syncthreads();
    float bv[8]; int bi[8];
#pragma unroll
    for (int r = 0; r < 8; ++r) { bv[r] = -3.4e38f; bi[r] = 0x7fffffff; }
    for (int nt = wave; nt < VOC / 16; nt += 16) {
      v8f acc = (v8f){0.f, 0.f, 0.f, 0.f, 0.f, 0.f, 0.f, 0.f};
      for (int k0 = 0; k0 < HD; k0 += 32) {
        v16h a = ld_frag(A16a + k0, HD, lane);
        v16h b = ld_frag(P.Wout + (size_t)nt * 16 * HD + k0, HD, lane);
        acc = __builtin_amdgcn_wmma_f32_16x16x32_f16(false, a, false, b,
                                                     (short)0, acc, false, false);
      }
      int n = nt * 16 + (lane & 15);
#pragma unroll
      for (int r = 0; r < 8; ++r)
        if (acc[r] > bv[r]) { bv[r] = acc[r]; bi[r] = n; }
    }
    // reduce argmax via LDS (overlay Q32, free here)
    float* candv = Q32;
    int*   candi = (int*)(Q32 + 16 * 256);
    {
      int mb = (lane >> 4) << 3;
      int slot = wave * 16 + (lane & 15);
#pragma unroll
      for (int r = 0; r < 8; ++r) {
        candv[(mb + r) * 256 + slot] = bv[r];
        candi[(mb + r) * 256 + slot] = bi[r];
      }
    }
    __syncthreads();
    if (tid < 16) {
      float best = -3.4e38f; int bidx = 0x7fffffff;
      for (int s = 0; s < 256; ++s) {
        float v = candv[tid * 256 + s];
        int  ix = candi[tid * 256 + s];
        if (v > best || (v == best && ix < bidx)) { best = v; bidx = ix; }
      }
      st_nxt[tid] = bidx;
      // ---- bookkeeping (matches reference ordering exactly) ----
      int r = tid;
      int done = st_done[r];
      int live = !done;
      int p = st_pos[r];
      int nxt = bidx;
      if (p < MAXTOK) {
        P.out_tok[(size_t)(b0 + r) * MAXTOK + p] = live ? (float)nxt : 0.f;
        P.out_gen[(size_t)(b0 + r) * MAXTOK + p] = live ? 1.f : 0.f;
      }
      p += live;
      int tip = st_tip[r] + live;
      int eos = (nxt == EOS_TOK) && (tip >= 1);
      int sw  = (eos || (tip >= MAXPER)) && live;
      int cur = st_cur[r] + sw;
      tip = sw ? 0 : tip;
      int sv = sw && (cur < KPH);
      int cp = (cur < KPH - 1) ? cur : (KPH - 1);
      if (sv) st_bounds[r * KPH + cp] = p;
      done = done || (cur >= KPH) || (sw && !st_act[r * KPH + cp]);
      st_pos[r] = p; st_tip[r] = tip; st_cur[r] = cur; st_done[r] = done;
      int mi = (cur < KPH - 1) ? cur : (KPH - 1);
      st_mi[r] = mi;
      st_mg[r] = (st_act[r * KPH + mi] && !done) ? 1.f : 0.f;
    }
    __syncthreads();
    if (t < MAXTOK - 1) {
      for (int i = tid; i < 16 * HD; i += 512) {
        int r = i >> 9, c = i & (HD - 1);
        X32[i] = P.token_emb[(size_t)st_nxt[r] * HD + c];
      }
      __syncthreads();
      attn_step(t + 1);
    }
  }
  if (tid < 16)
    for (int k = 0; k < KPH; ++k)
      P.out_bnd[(size_t)(b0 + tid) * KPH + k] = (float)st_bounds[tid * KPH + k];
}

// ---------------------------------------------------------------------------
extern "C" void kernel_launch(void* const* d_in, const int* in_sizes, int n_in,
                              void* d_out, int out_size, void* d_ws, size_t ws_size,
                              hipStream_t stream) {
  (void)in_sizes; (void)n_in; (void)out_size; (void)ws_size;
  const float* z_seq   = (const float*)d_in[0];
  const float* z_w     = (const float*)d_in[1];
  const float* W_l2d   = (const float*)d_in[2];
  const float* tok_emb = (const float*)d_in[3];
  const float* Wq  = (const float*)d_in[4];
  const float* Wk  = (const float*)d_in[5];
  const float* Wv  = (const float*)d_in[6];
  const float* Wo  = (const float*)d_in[7];
  const float* Wcq = (const float*)d_in[8];
  const float* Wck = (const float*)d_in[9];
  const float* Wcv = (const float*)d_in[10];
  const float* Wco = (const float*)d_in[11];
  const float* W1  = (const float*)d_in[12];
  const float* W2  = (const float*)d_in[13];
  const float* Wout = (const float*)d_in[14];

  char* base = (char*)d_ws;
  size_t off = 0;
  auto take = [&](size_t bytes) -> void* {
    void* p = base + off;
    off = (off + bytes + 255) & ~(size_t)255;
    return p;
  };
  h16* WqT  = (h16*)take((size_t)HD * HD * 2);
  h16* WkT  = (h16*)take((size_t)HD * HD * 2);
  h16* WvT  = (h16*)take((size_t)HD * HD * 2);
  h16* WoT  = (h16*)take((size_t)HD * HD * 2);
  h16* WcqT = (h16*)take((size_t)HD * HD * 2);
  h16* WckT = (h16*)take((size_t)HD * HD * 2);
  h16* WcvT = (h16*)take((size_t)HD * HD * 2);
  h16* WcoT = (h16*)take((size_t)HD * HD * 2);
  h16* W1T  = (h16*)take((size_t)HD * 4 * HD * 2);
  h16* W2T  = (h16*)take((size_t)4 * HD * HD * 2);
  h16* WoutT = (h16*)take((size_t)HD * VOC * 2);
  h16* memh  = (h16*)take((size_t)NB * KPH * NMEM * HD * 2);
  float* ckf = (float*)take((size_t)NB * KPH * NMEM * HD * 4);
  float* cvf = (float*)take((size_t)NB * KPH * NMEM * HD * 4);
  h16* kcache = (h16*)take((size_t)NB * TSLOT * HD * 2);
  h16* vcache = (h16*)take((size_t)NB * TSLOT * HD * 2);

  float* out_f  = (float*)d_out;
  float* out_tok = out_f;
  float* out_gen = out_f + (size_t)NB * MAXTOK;
  float* out_bnd = out_f + (size_t)2 * NB * MAXTOK;
  float* out_hid = out_f + (size_t)2 * NB * MAXTOK + (size_t)NB * KPH;

  // zero tokens/genm/bounds region (reference initializes them to zeros)
  int meta_n = 2 * NB * MAXTOK + NB * KPH;
  zero_kernel<<<(meta_n + 255) / 256, 256, 0, stream>>>(out_f, meta_n);

  auto tr = [&](const float* in, h16* o, int K, int N) {
    size_t n = (size_t)K * N;
    transpose_f16<<<(unsigned)((n + 255) / 256), 256, 0, stream>>>(in, o, K, N);
  };
  tr(Wq, WqT, HD, HD);   tr(Wk, WkT, HD, HD);   tr(Wv, WvT, HD, HD);
  tr(Wo, WoT, HD, HD);   tr(Wcq, WcqT, HD, HD); tr(Wck, WckT, HD, HD);
  tr(Wcv, WcvT, HD, HD); tr(Wco, WcoT, HD, HD);
  tr(W1, W1T, HD, 4 * HD);
  tr(W2, W2T, 4 * HD, HD);
  tr(Wout, WoutT, HD, VOC);

  mem_kernel<<<(NB * KPH * NMEM * HD) / 256, 256, 0, stream>>>(z_seq, z_w, W_l2d, memh);
  ckv_kernel<<<dim3(NB * KPH * NMEM / 16, 2), 256, 0, stream>>>(memh, WckT, WcvT, ckf, cvf);

  Params P;
  P.z_weights = z_w; P.token_emb = tok_emb;
  P.Wq = WqT; P.Wk = WkT; P.Wv = WvT; P.Wo = WoT;
  P.Wcq = WcqT; P.Wco = WcoT; P.W1 = W1T; P.W2 = W2T; P.Wout = WoutT;
  P.ckf = ckf; P.cvf = cvf; P.kc = kcache; P.vc = vcache;
  P.out_tok = out_tok; P.out_gen = out_gen; P.out_bnd = out_bnd; P.out_hid = out_hid;

  decode_kernel<<<NB / 16, 512, 0, stream>>>(P);
}
